// moe_70866960384125
// MI455X (gfx1250) — matrix operand order repaired and checked by reading, not compiled, not run on hardware
//
#include <hip/hip_runtime.h>
#include <hip/hip_bf16.h>
#include <stddef.h>

// ---------------------------------------------------------------------------
// Problem constants (from reference): B=8, S=2048, D=512, E=8, DEPTH=2, H=1024
// ---------------------------------------------------------------------------
#define Bq 8
#define Sq 2048
#define Dq 512
#define Eq 8
#define Hq 1024
#define TM 32            // tokens per workgroup tile
#define NTHREADS 256     // 8 waves of 32

typedef __bf16  v16bf __attribute__((ext_vector_type(16)));
typedef float   v8f   __attribute__((ext_vector_type(8)));
typedef unsigned int u32x4 __attribute__((ext_vector_type(4)));
typedef unsigned int u32x8 __attribute__((ext_vector_type(8)));

// float -> bf16 (round-to-nearest-even), stored as ushort
__device__ __forceinline__ unsigned short f2bf(float f) {
  unsigned int u = __builtin_bit_cast(unsigned int, f);
  u += 0x7FFFu + ((u >> 16) & 1u);
  return (unsigned short)(u >> 16);
}
__device__ __forceinline__ float bf2f(unsigned short u) {
  unsigned int w = ((unsigned int)u) << 16;
  return __builtin_bit_cast(float, w);
}

__device__ __forceinline__ v16bf pack16(u32x4 lo, u32x4 hi) {
  u32x8 w;
  w[0]=lo[0]; w[1]=lo[1]; w[2]=lo[2]; w[3]=lo[3];
  w[4]=hi[0]; w[5]=hi[1]; w[6]=hi[2]; w[7]=hi[3];
  return __builtin_bit_cast(v16bf, w);
}

// A fragment (16x32 bf16, row-major buffer, stride ldk elements).
// ISA 7.12.2: lane L holds row M=L&15; K = kb..kb+7 (v0-3) and kb+16..kb+23
// (v4-7) where kb = k0 + 8*(L>=16).
__device__ __forceinline__ v16bf load_A(const unsigned short* buf, int row0,
                                        int k0, int ldk, int lane) {
  int m  = lane & 15;
  int kb = k0 + ((lane >> 4) & 1) * 8;
  const unsigned short* p = buf + (size_t)(row0 + m) * ldk + kb;
  u32x4 lo = *(const u32x4*)p;          // K = kb .. kb+7
  u32x4 hi = *(const u32x4*)(p + 16);   // K = kb+16 .. kb+23
  return pack16(lo, hi);
}

// B fragment (32x16 bf16) from K-contiguous (transposed) weights Wt[n][k].
// ISA: lane L holds column N=L&15; K = ks..ks+7 and ks+16..ks+23 with ks = k0 + 8*(L>=16).
__device__ __forceinline__ v16bf load_B(const unsigned short* wt, int n0,
                                        int k0, int K, int lane) {
  int n  = lane & 15;
  int ks = k0 + ((lane >> 4) & 1) * 8;
  const unsigned short* p = wt + (size_t)(n0 + n) * K + ks;
  u32x4 lo = *(const u32x4*)p;          // K = ks .. ks+7
  u32x4 hi = *(const u32x4*)(p + 16);   // K = ks+16 .. ks+23
  return pack16(lo, hi);
}

#define WMMA_BF16(A, Bm, C) \
  __builtin_amdgcn_wmma_f32_16x16x32_bf16(false, (A), false, (Bm), (short)0, (C), false, false)

__device__ __forceinline__ float silu(float s) { return s / (1.f + __expf(-s)); }

// ---------------------------------------------------------------------------
// K0: transpose + fp32->bf16 convert.  src:[outer][R][C] -> dst:[outer][C][R]
// grid.x = outer*R (one block per source row; coalesced reads)
// ---------------------------------------------------------------------------
__global__ void tconv(const float* __restrict__ src, unsigned short* __restrict__ dst,
                      int R, int C) {
  int o = blockIdx.x / R;
  int r = blockIdx.x - o * R;
  const float* s = src + ((size_t)o * R + r) * C;
  for (int c = threadIdx.x; c < C; c += blockDim.x)
    dst[((size_t)o * C + c) * R + r] = f2bf(s[c]);
}

// ---------------------------------------------------------------------------
// K1: scores[b,s] = silu(x @ att_w1 + b1) @ att_w2 + b2     (WMMA tile GEMM)
// ---------------------------------------------------------------------------
__global__ __launch_bounds__(NTHREADS) void scores_kernel(
    const float* __restrict__ x, const unsigned short* __restrict__ AW1t,
    const float* __restrict__ att_b1, const float* __restrict__ att_w2,
    const float* __restrict__ att_b2, float* __restrict__ scores) {
  __shared__ __align__(16) unsigned short xs[TM][Dq + 8];
  __shared__ float us[TM][Dq];
  int tid = threadIdx.x, lane = tid & 31, wv = tid >> 5;
  int tok0 = blockIdx.x * TM;

  for (int i = tid; i < TM * Dq; i += NTHREADS) {
    int r = i >> 9, c = i & (Dq - 1);
    xs[r][c] = f2bf(x[(size_t)(tok0 + r) * Dq + c]);
  }
  __syncthreads();

  int mb = ((lane >> 4) & 1) * 8, n = lane & 15;
#pragma unroll
  for (int t2 = 0; t2 < 4; ++t2) {          // 8 waves * 4 = 32 col-tiles (512)
    int n0 = (wv + 8 * t2) * 16;
    float bb = att_b1[n0 + n];
    v8f a0, a1;
#pragma unroll
    for (int r = 0; r < 8; ++r) { a0[r] = bb; a1[r] = bb; }
    for (int k0 = 0; k0 < Dq; k0 += 32) {
      v16bf bm  = load_B(AW1t, n0, k0, Dq, lane);
      v16bf am0 = load_A(&xs[0][0], 0,  k0, Dq + 8, lane);
      v16bf am1 = load_A(&xs[0][0], 16, k0, Dq + 8, lane);
      a0 = WMMA_BF16(am0, bm, a0);
      a1 = WMMA_BF16(am1, bm, a1);
    }
#pragma unroll
    for (int r = 0; r < 8; ++r) {
      us[mb + r][n0 + n]      = silu(a0[r]);
      us[16 + mb + r][n0 + n] = silu(a1[r]);
    }
  }
  __syncthreads();
  if (tid < TM) {
    float s = att_b2[0];
    for (int k = 0; k < Dq; ++k) s += us[tid][k] * att_w2[k];
    scores[tok0 + tid] = s;
  }
}

// ---------------------------------------------------------------------------
// K2: x_att[b,d] = sum_s scores[b,s] * x[b,s,d]      grid = B, 256 threads
// ---------------------------------------------------------------------------
__global__ __launch_bounds__(NTHREADS) void xatt_kernel(
    const float* __restrict__ x, const float* __restrict__ scores,
    float* __restrict__ x_att) {
  __shared__ float ssc[Sq];
  int b = blockIdx.x, tid = threadIdx.x;
  for (int i = tid; i < Sq; i += NTHREADS) ssc[i] = scores[b * Sq + i];
  __syncthreads();
  float a0 = 0.f, a1 = 0.f;
  const float* xb = x + (size_t)b * Sq * Dq;
  for (int s = 0; s < Sq; ++s) {
    float sc = ssc[s];
    a0 += sc * xb[(size_t)s * Dq + tid];
    a1 += sc * xb[(size_t)s * Dq + tid + NTHREADS];
  }
  x_att[b * Dq + tid] = a0;
  x_att[b * Dq + tid + NTHREADS] = a1;
}

// ---------------------------------------------------------------------------
// K3: gates gw1,gw2 [B,E]  (tiny; one workgroup)
// ---------------------------------------------------------------------------
__global__ __launch_bounds__(NTHREADS) void gates_kernel(
    const float* __restrict__ xatt,
    const float* __restrict__ w1a, const float* __restrict__ b1a,
    const float* __restrict__ w2a, const float* __restrict__ b2a,
    const float* __restrict__ w1b, const float* __restrict__ b1b,
    const float* __restrict__ w2b, const float* __restrict__ b2b,
    float* __restrict__ gw1, float* __restrict__ gw2) {
  __shared__ float v[Dq];
  int tid = threadIdx.x;
  for (int b = 0; b < Bq; ++b) {
    for (int pass = 0; pass < 2; ++pass) {
      const float* w1 = pass ? w1b : w1a;  const float* b1 = pass ? b1b : b1a;
      const float* w2 = pass ? w2b : w2a;  const float* b2 = pass ? b2b : b2a;
      float* gout = pass ? gw2 : gw1;
      for (int nn = tid; nn < Dq; nn += NTHREADS) {
        float s = b1[nn];
        for (int k = 0; k < Dq; ++k) s += xatt[b * Dq + k] * w1[(size_t)k * Dq + nn];
        v[nn] = silu(s);
      }
      __syncthreads();
      if (tid < Eq) {
        float s = b2[tid];
        for (int k = 0; k < Dq; ++k) s += v[k] * w2[(size_t)k * Eq + tid];
        gout[b * Eq + tid] = s;
      }
      __syncthreads();
    }
  }
}

__global__ void zero_kernel(float* p, int n) {
  int i = blockIdx.x * blockDim.x + threadIdx.x;
  if (i < n) p[i] = 0.f;
}

// ---------------------------------------------------------------------------
// K4: fused soft-MoE.  Per 32-token tile: loop experts x depth, H stays in
// LDS, gate-weighted feature1 accumulated in registers, written once.
// ---------------------------------------------------------------------------
__global__ __launch_bounds__(NTHREADS) void moe_main(
    const float* __restrict__ x, const unsigned short* __restrict__ W1t,
    const unsigned short* __restrict__ W2t, const float* __restrict__ eb1,
    const float* __restrict__ eb2, const float* __restrict__ gw2v,
    float* __restrict__ out1) {
  __shared__ __align__(16) unsigned short xs0[TM][Dq + 8];   // original x (bf16)
  __shared__ __align__(16) unsigned short xs1[TM][Dq + 8];   // depth intermediate
  __shared__ __align__(16) unsigned short hsb[TM][Hq + 8];   // hidden (bf16)

  int tid = threadIdx.x, lane = tid & 31, wv = tid >> 5;
  int tok0 = blockIdx.x * TM;
  int b = tok0 >> 11;                      // S = 2048, TM divides S
  int mb = ((lane >> 4) & 1) * 8, n = lane & 15;

  for (int i = tid; i < TM * Dq; i += NTHREADS) {
    int r = i >> 9, c = i & (Dq - 1);
    xs0[r][c] = f2bf(x[(size_t)(tok0 + r) * Dq + c]);
  }
  __syncthreads();

  v8f facc[8];
#pragma unroll
  for (int t = 0; t < 8; ++t)
#pragma unroll
    for (int r = 0; r < 8; ++r) facc[t][r] = 0.f;

  for (int e = 0; e < Eq; ++e) {
    float gate = gw2v[b * Eq + e];
    for (int j = 0; j < 2; ++j) {
      const unsigned short(*cur)[Dq + 8] = j ? xs1 : xs0;
      // ---- GEMM1: h[TM,H] = cur @ W1 + eb1 --------------------------------
      const unsigned short* w1 = W1t + (size_t)(e * 2 + j) * Hq * Dq;
      const float* bias1 = eb1 + (size_t)(e * 2 + j) * Hq;
#pragma unroll
      for (int t2 = 0; t2 < 8; ++t2) {     // 8 waves * 8 = 64 col-tiles (1024)
        int n0 = (wv + 8 * t2) * 16;
        float bb = bias1[n0 + n];
        v8f a0, a1;
#pragma unroll
        for (int r = 0; r < 8; ++r) { a0[r] = bb; a1[r] = bb; }
        for (int k0 = 0; k0 < Dq; k0 += 32) {
          v16bf bm  = load_B(w1, n0, k0, Dq, lane);
          v16bf am0 = load_A(&cur[0][0], 0,  k0, Dq + 8, lane);
          v16bf am1 = load_A(&cur[0][0], 16, k0, Dq + 8, lane);
          a0 = WMMA_BF16(am0, bm, a0);
          a1 = WMMA_BF16(am1, bm, a1);
        }
#pragma unroll
        for (int r = 0; r < 8; ++r) {
          hsb[mb + r][n0 + n]      = f2bf(a0[r]);
          hsb[16 + mb + r][n0 + n] = f2bf(a1[r]);
        }
      }
      __syncthreads();
      // ---- GEMM2: y[TM,D] = h @ W2 + eb2 ----------------------------------
      const unsigned short* w2 = W2t + (size_t)(e * 2 + j) * Dq * Hq;
      const float* bias2 = eb2 + (size_t)(e * 2 + j) * Dq;
#pragma unroll
      for (int t2 = 0; t2 < 4; ++t2) {     // 8 waves * 4 = 32 col-tiles (512)
        int n0 = (wv + 8 * t2) * 16;
        float bb = bias2[n0 + n];
        v8f a0, a1;
#pragma unroll
        for (int r = 0; r < 8; ++r) { a0[r] = bb; a1[r] = bb; }
        for (int k0 = 0; k0 < Hq; k0 += 32) {
          v16bf bm  = load_B(w2, n0, k0, Hq, lane);
          v16bf am0 = load_A(&hsb[0][0], 0,  k0, Hq + 8, lane);
          v16bf am1 = load_A(&hsb[0][0], 16, k0, Hq + 8, lane);
          a0 = WMMA_BF16(am0, bm, a0);
          a1 = WMMA_BF16(am1, bm, a1);
        }
        if (j == 0) {
#pragma unroll
          for (int r = 0; r < 8; ++r) {
            xs1[mb + r][n0 + n]      = f2bf(a0[r]);
            xs1[16 + mb + r][n0 + n] = f2bf(a1[r]);
          }
        } else {
#pragma unroll
          for (int r = 0; r < 8; ++r) {
            facc[t2 * 2][r]     += gate * a0[r];
            facc[t2 * 2 + 1][r] += gate * a1[r];
          }
        }
      }
      __syncthreads();
    }
  }
  // feature1 store (fp32), one pass
#pragma unroll
  for (int t2 = 0; t2 < 4; ++t2) {
    int n0 = (wv + 8 * t2) * 16;
#pragma unroll
    for (int r = 0; r < 8; ++r) {
      out1[(size_t)(tok0 + mb + r) * Dq + n0 + n]      = facc[t2 * 2][r];
      out1[(size_t)(tok0 + 16 + mb + r) * Dq + n0 + n] = facc[t2 * 2 + 1][r];
    }
  }
}

// ---------------------------------------------------------------------------
// K5: feature[:,0] = sum_e gw1[b,e] * expert_e(x[b,0,:])   grid = B*E
// ---------------------------------------------------------------------------
__global__ __launch_bounds__(NTHREADS) void feature_kernel(
    const float* __restrict__ x, const unsigned short* __restrict__ W1t,
    const unsigned short* __restrict__ W2t, const float* __restrict__ eb1,
    const float* __restrict__ eb2, const float* __restrict__ gw1v,
    float* __restrict__ feat) {
  int b = blockIdx.x >> 3, e = blockIdx.x & 7;
  int tid = threadIdx.x;
  __shared__ float v[Dq];
  __shared__ float h[Hq];
  for (int i = tid; i < Dq; i += NTHREADS) v[i] = x[(size_t)b * Sq * Dq + i];
  __syncthreads();
  for (int j = 0; j < 2; ++j) {
    const unsigned short* w1 = W1t + (size_t)(e * 2 + j) * Hq * Dq;
    for (int nn = tid; nn < Hq; nn += NTHREADS) {
      float s = eb1[(size_t)(e * 2 + j) * Hq + nn];
      const unsigned short* wr = w1 + (size_t)nn * Dq;
      float s0 = 0, s1 = 0, s2 = 0, s3 = 0;
      for (int k = 0; k < Dq; k += 4) {
        s0 += v[k]     * bf2f(wr[k]);
        s1 += v[k + 1] * bf2f(wr[k + 1]);
        s2 += v[k + 2] * bf2f(wr[k + 2]);
        s3 += v[k + 3] * bf2f(wr[k + 3]);
      }
      h[nn] = s + (s0 + s1) + (s2 + s3);
    }
    __syncthreads();
    const unsigned short* w2 = W2t + (size_t)(e * 2 + j) * Dq * Hq;
    for (int nn = tid; nn < Dq; nn += NTHREADS) {
      float s = eb2[(size_t)(e * 2 + j) * Dq + nn];
      const unsigned short* wr = w2 + (size_t)nn * Hq;
      float s0 = 0, s1 = 0, s2 = 0, s3 = 0;
      for (int k = 0; k < Hq; k += 4) {
        s0 += h[k]     * bf2f(wr[k]);
        s1 += h[k + 1] * bf2f(wr[k + 1]);
        s2 += h[k + 2] * bf2f(wr[k + 2]);
        s3 += h[k + 3] * bf2f(wr[k + 3]);
      }
      v[nn] = s + (s0 + s1) + (s2 + s3);
    }
    __syncthreads();
  }
  float g = gw1v[b * Eq + e];
  for (int i = tid; i < Dq; i += NTHREADS) atomicAdd(&feat[b * Dq + i], g * v[i]);
}

// ---------------------------------------------------------------------------
// Host launch
// ---------------------------------------------------------------------------
extern "C" void kernel_launch(void* const* d_in, const int* in_sizes, int n_in,
                              void* d_out, int out_size, void* d_ws, size_t ws_size,
                              hipStream_t stream) {
  const float* x      = (const float*)d_in[0];
  const float* att_w1 = (const float*)d_in[1];
  const float* att_b1 = (const float*)d_in[2];
  const float* att_w2 = (const float*)d_in[3];
  const float* att_b2 = (const float*)d_in[4];
  const float* g1_w1  = (const float*)d_in[5];
  const float* g1_b1  = (const float*)d_in[6];
  const float* g1_w2  = (const float*)d_in[7];
  const float* g1_b2  = (const float*)d_in[8];
  const float* g2_w1  = (const float*)d_in[9];
  const float* g2_b1  = (const float*)d_in[10];
  const float* g2_w2  = (const float*)d_in[11];
  const float* g2_b2  = (const float*)d_in[12];
  const float* ew1    = (const float*)d_in[13];
  const float* eb1    = (const float*)d_in[14];
  const float* ew2    = (const float*)d_in[15];
  const float* eb2    = (const float*)d_in[16];

  char* ws = (char*)d_ws;
  const size_t OFF_W1T    = 0;                              // 16 MB bf16
  const size_t OFF_W2T    = (size_t)16 << 20;               // 16 MB bf16
  const size_t OFF_AW1T   = (size_t)32 << 20;               // 512 KB bf16
  const size_t OFF_SCORES = OFF_AW1T + 524288;              // 64 KB fp32
  const size_t OFF_XATT   = OFF_SCORES + 65536;             // 16 KB fp32
  const size_t OFF_GW1    = OFF_XATT + 16384;
  const size_t OFF_GW2    = OFF_GW1 + 256;

  unsigned short* W1t  = (unsigned short*)(ws + OFF_W1T);   // [E*2][H][D]
  unsigned short* W2t  = (unsigned short*)(ws + OFF_W2T);   // [E*2][D][H]
  unsigned short* AW1t = (unsigned short*)(ws + OFF_AW1T);  // [D][D]
  float* scores = (float*)(ws + OFF_SCORES);
  float* x_att  = (float*)(ws + OFF_XATT);
  float* gw1    = (float*)(ws + OFF_GW1);
  float* gw2    = (float*)(ws + OFF_GW2);
  float* out    = (float*)d_out;                            // [B*D | B*S*D]

  // weight prep (bf16, K-contiguous per output column)
  tconv<<<Eq * 2 * Dq, NTHREADS, 0, stream>>>(ew1, W1t, Dq, Hq);   // 16*512 rows
  tconv<<<Eq * 2 * Hq, NTHREADS, 0, stream>>>(ew2, W2t, Hq, Dq);   // 16*1024 rows
  tconv<<<Dq, NTHREADS, 0, stream>>>(att_w1, AW1t, Dq, Dq);

  // token attention + pooling + gates
  scores_kernel<<<(Bq * Sq) / TM, NTHREADS, 0, stream>>>(x, AW1t, att_b1, att_w2,
                                                         att_b2, scores);
  xatt_kernel<<<Bq, NTHREADS, 0, stream>>>(x, scores, x_att);
  gates_kernel<<<1, NTHREADS, 0, stream>>>(x_att, g1_w1, g1_b1, g1_w2, g1_b2,
                                           g2_w1, g2_b1, g2_w2, g2_b2, gw1, gw2);

  // outputs
  zero_kernel<<<(Bq * Dq + NTHREADS - 1) / NTHREADS, NTHREADS, 0, stream>>>(out, Bq * Dq);
  moe_main<<<(Bq * Sq) / TM, NTHREADS, 0, stream>>>(x, W1t, W2t, eb1, eb2, gw2,
                                                    out + Bq * Dq);
  feature_kernel<<<Bq * Eq, NTHREADS, 0, stream>>>(x, W1t, W2t, eb1, eb2, gw1, out);
}
